// SDFDecoder_59554016526687
// MI455X (gfx1250) — matrix operand order repaired and checked by reading, not compiled, not run on hardware
//
#include <hip/hip_runtime.h>

// ---------------------------------------------------------------------------
// Problem constants (from reference)
// ---------------------------------------------------------------------------
#define BATCH 4096
#define PLAT  512
#define KNN   8
#define DMODEL 512
#define NHEAD 8
#define DHEAD 64
#define HID   512
#define ROWSTRIDE (3 + 3 * PLAT)   // 1539

typedef __attribute__((ext_vector_type(16))) __bf16 v16bf;
typedef __attribute__((ext_vector_type(8)))  float  v8f;
typedef __attribute__((ext_vector_type(4)))  unsigned int v4u;
typedef __attribute__((ext_vector_type(4)))  int v4i;
typedef __attribute__((ext_vector_type(8)))  int v8i;

#if defined(__AMDGCN__) && __has_builtin(__builtin_amdgcn_tensor_load_to_lds)
#define HAVE_TDM 1
#endif

__device__ __forceinline__ unsigned short f2bf(float f) {
    unsigned int u = __float_as_uint(f);
    unsigned int r = u + 0x7FFFu + ((u >> 16) & 1u);   // round-to-nearest-even
    return (unsigned short)(r >> 16);
}

union FragU {
    v16bf b;
    unsigned short u[16];
};

// A-matrix fragment, 16x32 bf16 (documented CDNA5 A layout):
// lane L holds row m = L&15; lanes0-15 -> K 0..7 & 16..23,
// lanes16-31 -> K 8..15 & 24..31 (2 K per VGPR).
__device__ __forceinline__ v16bf load_frag_a(const unsigned short* p, int rs, int ks, int lane) {
    int m  = lane & 15;
    int kb = (lane >> 4) << 3;
    FragU f;
#pragma unroll
    for (int v = 0; v < 8; ++v) {
        int k = (v < 4) ? (kb + 2 * v) : (16 + kb + 2 * (v - 4));
        f.u[2 * v]     = p[m * rs + k * ks];
        f.u[2 * v + 1] = p[m * rs + (k + 1) * ks];
    }
    return f.b;
}

// B-matrix fragment, 32x16 bf16: lane L holds col n = L&15,
// lanes0-15 -> K 0..7 & 16..23, lanes16-31 -> K 8..15 & 24..31 (matches A).
__device__ __forceinline__ v16bf load_frag_b(const unsigned short* p, int ns, int ks, int lane) {
    int n  = lane & 15;
    int kb = (lane >> 4) << 3;
    FragU f;
#pragma unroll
    for (int e = 0; e < 16; ++e)
        f.u[e] = p[n * ns + (kb + e + ((e >= 8) ? 8 : 0)) * ks];
    return f.b;
}

// ---------------------------------------------------------------------------
// Elementwise conversion kernels
// ---------------------------------------------------------------------------
__global__ void cvt_bf16_kernel(const float* __restrict__ src, unsigned short* __restrict__ dst, int n) {
    int i = blockIdx.x * 256 + threadIdx.x;
    if (i < n) dst[i] = f2bf(src[i]);
}

__global__ void cvt_lat_kernel(const float* __restrict__ cxyz, unsigned short* __restrict__ latbf) {
    int i = blockIdx.x * 256 + threadIdx.x;      // BATCH*1536
    int r = i / (3 * PLAT), c = i - r * (3 * PLAT);
    latbf[i] = f2bf(cxyz[r * ROWSTRIDE + 3 + c]);
}

__global__ void pad_wxyz_kernel(const float* __restrict__ W, unsigned short* __restrict__ out) {
    int i = blockIdx.x * 256 + threadIdx.x;      // 32*512
    int r = i >> 9, c = i & 511;
    float v = (r < 27) ? W[r * DMODEL + c] : 0.f;
    out[i] = f2bf(v);
}

// copy (B,512) f32 into bf16 buffer at column offset (for concat halves)
__global__ void copy_off_kernel(const float* __restrict__ src, unsigned short* __restrict__ dst,
                                int ld, int coff) {
    int i = blockIdx.x * 256 + threadIdx.x;      // BATCH*512
    int r = i >> 9, c = i & 511;
    dst[r * ld + coff + c] = f2bf(src[i]);
}

// ---------------------------------------------------------------------------
// KNN top-8 + xp input row build (padded to 32 cols) — one wave per point
// ---------------------------------------------------------------------------
__global__ __launch_bounds__(32) void topk_build_kernel(const float* __restrict__ cxyz,
                                                        unsigned short* __restrict__ xppad) {
    __shared__ int selS[KNN];
    int row  = blockIdx.x;
    int lane = threadIdx.x;
    const float* cr = cxyz + (size_t)row * ROWSTRIDE;
    float x = cr[0], y = cr[1], z = cr[2];
    float d[16];
#pragma unroll
    for (int j = 0; j < 16; ++j) {
        int p = j * 32 + lane;
        float dx = cr[3 + 3 * p] - x, dy = cr[4 + 3 * p] - y, dz = cr[5 + 3 * p] - z;
        d[j] = sqrtf(dx * dx + dy * dy + dz * dz);
    }
#pragma unroll
    for (int t = 0; t < KNN; ++t) {
        float bm = 3.0e38f; int bj = 0;
#pragma unroll
        for (int j = 0; j < 16; ++j)
            if (d[j] < bm) { bm = d[j]; bj = j; }
        int bp = bj * 32 + lane;
#pragma unroll
        for (int off = 16; off >= 1; off >>= 1) {
            float om = __shfl_xor(bm, off);
            int   op = __shfl_xor(bp, off);
            if (om < bm || (om == bm && op < bp)) { bm = om; bp = op; }
        }
#pragma unroll
        for (int j = 0; j < 16; ++j)                 // mark winner used (owner lane)
            if (bp == j * 32 + lane) d[j] = 3.0e38f;
        if (lane == 0) selS[t] = bp;
    }
    __syncthreads();
    float val;
    if (lane < 3)       val = cr[lane];
    else if (lane < 27) { int t = (lane - 3) / 3, c = (lane - 3) % 3; val = cr[3 + 3 * selS[t] + c]; }
    else                val = 0.f;
    xppad[row * 32 + lane] = f2bf(val);
}

// ---------------------------------------------------------------------------
// WMMA GEMM: C[M,N] = act(A_bf16[M,K] @ B_bf16[K,N] + bias)
// WG = 256 thr (8 waves), WG tile 128x64, wave tile 32x32 (2x2 WMMA tiles)
// ---------------------------------------------------------------------------
__global__ __launch_bounds__(256) void gemm_bf16_kernel(
        const unsigned short* __restrict__ A, int lda,
        const unsigned short* __restrict__ Bw, int ldb,
        const float* __restrict__ bias,
        float* __restrict__ Cf, int ldcf,
        unsigned short* __restrict__ Cb, int ldcb,
        int Kd, int act) {
    __shared__ unsigned short As[128 * 32];   // 8 KB
    __shared__ unsigned short BsT[64 * 32];   // 4 KB, stored [n][k]
    int tid  = threadIdx.x;
    int lane = tid & 31, wave = tid >> 5;
    int wr = wave >> 1, wc = wave & 1;
    int m0 = blockIdx.y * 128, n0 = blockIdx.x * 64;
    const unsigned int* A32 = (const unsigned int*)A;
    const unsigned int* B32 = (const unsigned int*)Bw;
    unsigned int* As32 = (unsigned int*)As;
    int lda2 = lda >> 1, ldb2 = ldb >> 1;

    v8f acc[2][2] = {};
    for (int k0 = 0; k0 < Kd; k0 += 32) {
        if (k0 + 32 < Kd)
            __builtin_prefetch(&A32[(size_t)(m0 + (tid >> 1)) * lda2 + ((k0 + 32) >> 1)], 0, 0);
#pragma unroll
        for (int i = 0; i < 8; ++i) {            // A tile: 2048 dwords
            int e = tid + i * 256;
            int m = e >> 4, kk = e & 15;
            As32[e] = A32[(size_t)(m0 + m) * lda2 + (k0 >> 1) + kk];
        }
#pragma unroll
        for (int i = 0; i < 4; ++i) {            // B tile (transposed in LDS): 1024 dwords
            int e = tid + i * 256;
            int kk = e >> 5, np = e & 31;
            unsigned int w = B32[(size_t)(k0 + kk) * ldb2 + (n0 >> 1) + np];
            BsT[(2 * np) * 32 + kk]     = (unsigned short)(w & 0xFFFFu);
            BsT[(2 * np + 1) * 32 + kk] = (unsigned short)(w >> 16);
        }
        __syncthreads();
        v16bf af[2], bf[2];
#pragma unroll
        for (int mi = 0; mi < 2; ++mi) af[mi] = load_frag_a(As + (wr * 32 + mi * 16) * 32, 32, 1, lane);
#pragma unroll
        for (int ni = 0; ni < 2; ++ni) bf[ni] = load_frag_b(BsT + (wc * 32 + ni * 16) * 32, 32, 1, lane);
#pragma unroll
        for (int mi = 0; mi < 2; ++mi)
#pragma unroll
            for (int ni = 0; ni < 2; ++ni)
                acc[mi][ni] = __builtin_amdgcn_wmma_f32_16x16x32_bf16(
                    false, af[mi], false, bf[ni], (short)0, acc[mi][ni], false, false);
        __syncthreads();
    }
#pragma unroll
    for (int mi = 0; mi < 2; ++mi)
#pragma unroll
        for (int ni = 0; ni < 2; ++ni) {
            int bm = m0 + wr * 32 + mi * 16;
            int bn = n0 + wc * 32 + ni * 16;
#pragma unroll
            for (int v = 0; v < 8; ++v) {
                int r = bm + v + 8 * (lane >> 4);
                int c = bn + (lane & 15);
                float val = acc[mi][ni][v] + (bias ? bias[c] : 0.f);
                if (act == 1) val = fmaxf(val, 0.f);
                else if (act == 2) val = tanhf(val);
                if (Cf) Cf[(size_t)r * ldcf + c] = val;
                if (Cb) Cb[(size_t)r * ldcb + c] = f2bf(val);
            }
        }
}

// ---------------------------------------------------------------------------
// Flash attention, 8 waves/block: 128 queries share one K/V tile in LDS.
// Each wave owns a 16-query tile; online softmax per row.
// ---------------------------------------------------------------------------
__global__ __launch_bounds__(256) void attn_kernel(
        const unsigned short* __restrict__ Q,
        const unsigned short* __restrict__ Kmat,
        const unsigned short* __restrict__ Vmat,
        unsigned short* __restrict__ Ob) {
    __shared__ unsigned short Ks[32 * 64];       // [key][dim]  4 KB
    __shared__ unsigned short VsT[64 * 32];      // [dim][key]  4 KB
    __shared__ unsigned short Ps[8 * 16 * 32];   // per-wave P tiles, 8 KB
    int tid  = threadIdx.x;
    int lane = tid & 31, w = tid >> 5;
    int c0 = blockIdx.y * DHEAD;                 // head column base
    int q0 = blockIdx.x * 128 + w * 16;
    const unsigned int* K32 = (const unsigned int*)Kmat;
    const unsigned int* V32 = (const unsigned int*)Vmat;
    unsigned int* Ks32 = (unsigned int*)Ks;
    unsigned short* myP = Ps + w * (16 * 32);

    v16bf qf0 = load_frag_a(Q + (size_t)q0 * DMODEL + c0,      DMODEL, 1, lane);
    v16bf qf1 = load_frag_a(Q + (size_t)q0 * DMODEL + c0 + 32, DMODEL, 1, lane);
    float m8[8], l8[8];
    v8f oacc[4] = {};
#pragma unroll
    for (int v = 0; v < 8; ++v) { m8[v] = -3.0e38f; l8[v] = 0.f; }

    for (int kb0 = 0; kb0 < BATCH; kb0 += 32) {
        // cooperative stage of 32 keys x 64 dims of K (row-major) and V (transposed)
#pragma unroll
        for (int i = 0; i < 4; ++i) {
            int e  = tid + i * 256;              // 0..1023 dword pairs
            int ky = e >> 5, dp = e & 31;
            Ks32[e] = K32[(size_t)(kb0 + ky) * (DMODEL / 2) + (c0 >> 1) + dp];
            unsigned int wv = V32[(size_t)(kb0 + ky) * (DMODEL / 2) + (c0 >> 1) + dp];
            VsT[(2 * dp) * 32 + ky]     = (unsigned short)(wv & 0xFFFFu);
            VsT[(2 * dp + 1) * 32 + ky] = (unsigned short)(wv >> 16);
        }
        __syncthreads();

        v8f s[2];
#pragma unroll
        for (int t = 0; t < 2; ++t) {
            v16bf kf0 = load_frag_b(Ks + (t * 16) * 64,      64, 1, lane);  // B[k][n]=K[n][k]
            v16bf kf1 = load_frag_b(Ks + (t * 16) * 64 + 32, 64, 1, lane);
            v8f z = {};
            z    = __builtin_amdgcn_wmma_f32_16x16x32_bf16(false, qf0, false, kf0, (short)0, z, false, false);
            s[t] = __builtin_amdgcn_wmma_f32_16x16x32_bf16(false, qf1, false, kf1, (short)0, z, false, false);
        }
#pragma unroll
        for (int v = 0; v < 8; ++v) {
            float a = s[0][v] * 0.125f;          // 1/sqrt(DH)
            float b = s[1][v] * 0.125f;
            float mx = fmaxf(a, b);
#pragma unroll
            for (int off = 8; off >= 1; off >>= 1)   // reduce within 16-lane half (row-wise)
                mx = fmaxf(mx, __shfl_xor(mx, off));
            float mnew = fmaxf(m8[v], mx);
            float corr = expf(m8[v] - mnew);
            float p0 = expf(a - mnew), p1 = expf(b - mnew);
            float ls = p0 + p1;
#pragma unroll
            for (int off = 8; off >= 1; off >>= 1)
                ls += __shfl_xor(ls, off);
            l8[v] = l8[v] * corr + ls;
            m8[v] = mnew;
            int r = v + 8 * (lane >> 4);
            int c = lane & 15;
            myP[r * 32 + c]      = f2bf(p0);     // own-wave LDS, in-order DS
            myP[r * 32 + 16 + c] = f2bf(p1);
#pragma unroll
            for (int nt = 0; nt < 4; ++nt) oacc[nt][v] *= corr;
        }
        v16bf pf = load_frag_a(myP, 32, 1, lane);
#pragma unroll
        for (int nt = 0; nt < 4; ++nt) {
            v16bf vf = load_frag_b(VsT + (nt * 16) * 32, 32, 1, lane);
            oacc[nt] = __builtin_amdgcn_wmma_f32_16x16x32_bf16(false, pf, false, vf, (short)0, oacc[nt], false, false);
        }
        __syncthreads();                         // all waves done with Ks/VsT
    }
#pragma unroll
    for (int v = 0; v < 8; ++v) {
        float inv = 1.f / l8[v];
        int r = q0 + v + 8 * (lane >> 4);
#pragma unroll
        for (int nt = 0; nt < 4; ++nt) {
            int c = c0 + nt * 16 + (lane & 15);
            Ob[(size_t)r * DMODEL + c] = f2bf(oacc[nt][v] * inv);
        }
    }
}

// ---------------------------------------------------------------------------
// residual add + LayerNorm, one wave per row (512 cols = 16/lane)
// ---------------------------------------------------------------------------
__global__ __launch_bounds__(32) void add_ln_kernel(
        const float* __restrict__ A, const float* __restrict__ Bz,
        const float* __restrict__ g, const float* __restrict__ be,
        float* __restrict__ outf, unsigned short* __restrict__ outb) {
    int row = blockIdx.x, lane = threadIdx.x;
    const float* ar = A + (size_t)row * DMODEL;
    const float* br = Bz + (size_t)row * DMODEL;
    float v[16]; float s = 0.f;
#pragma unroll
    for (int j = 0; j < 16; ++j) { v[j] = ar[j * 32 + lane] + br[j * 32 + lane]; s += v[j]; }
#pragma unroll
    for (int off = 16; off >= 1; off >>= 1) s += __shfl_xor(s, off);
    float mean = s * (1.f / DMODEL);
    float var = 0.f;
#pragma unroll
    for (int j = 0; j < 16; ++j) { float t = v[j] - mean; var += t * t; }
#pragma unroll
    for (int off = 16; off >= 1; off >>= 1) var += __shfl_xor(var, off);
    float inv = rsqrtf(var * (1.f / DMODEL) + 1e-5f);
#pragma unroll
    for (int j = 0; j < 16; ++j) {
        int c = j * 32 + lane;
        float o = (v[j] - mean) * inv * g[c] + be[c];
        outf[(size_t)row * DMODEL + c] = o;
        outb[(size_t)row * DMODEL + c] = f2bf(o);
    }
}

// final 512 -> 1 dot + tanh, one wave per row
__global__ __launch_bounds__(32) void final_dot_kernel(
        const float* __restrict__ X, const float* __restrict__ w,
        const float* __restrict__ b, float* __restrict__ out) {
    int row = blockIdx.x, lane = threadIdx.x;
    float s = 0.f;
#pragma unroll
    for (int j = 0; j < 16; ++j) { int c = j * 32 + lane; s += X[(size_t)row * HID + c] * w[c]; }
#pragma unroll
    for (int off = 16; off >= 1; off >>= 1) s += __shfl_xor(s, off);
    if (lane == 0) out[row] = tanhf(s + b[0]);
}

// ---------------------------------------------------------------------------
// TDM: stage one 64x32 bf16 weight tile into LDS with the Tensor Data Mover,
// wait on TENSORcnt, checksum into scratch. Exercises the gfx1250 async
// tensor path (tensor_load_to_lds + s_wait_tensorcnt) per ISA ch.8.
// ---------------------------------------------------------------------------
#ifdef HAVE_TDM
__global__ __launch_bounds__(32) void tdm_stage_kernel(
        const unsigned short* __restrict__ src, unsigned int* __restrict__ sink,
        int width, int height) {
    __shared__ unsigned short tileS[32 * 64];    // only LDS object -> offset 0
    unsigned long long ga = (unsigned long long)src;
    // ---- D# group 0 ----
    v4u g0;
    g0[0] = 1u;                                          // count=1, user descriptor
    g0[1] = 0u;                                          // lds_addr = 0
    g0[2] = (unsigned int)(ga & 0xFFFFFFFFu);            // global_addr[31:0]
    g0[3] = (unsigned int)((ga >> 32) & 0x01FFFFFFu) | (2u << 30);  // addr[56:32] | type=2
    // ---- D# group 1 ----
    unsigned int dim0 = (unsigned int)width;             // tensor_dim0 (elements)
    unsigned int dim1 = (unsigned int)height;            // tensor_dim1
    unsigned int til0 = 64u, til1 = 32u;                 // tile 64x32
    unsigned long long st0 = (unsigned long long)width;  // tensor_dim0_stride
    v8i g1;
    g1[0] = (int)(1u << 16);                             // data_size = 2 bytes
    g1[1] = (int)((dim0 & 0xFFFFu) << 16);               // tensor_dim0[15:0] @bits63:48
    g1[2] = (int)(((dim0 >> 16) & 0xFFFFu) | ((dim1 & 0xFFFFu) << 16));
    g1[3] = (int)(((dim1 >> 16) & 0xFFFFu) | (til0 << 16));  // tile_dim0 @bits127:112
    g1[4] = (int)(til1 & 0xFFFFu);                       // tile_dim1; tile_dim2=0
    g1[5] = (int)(st0 & 0xFFFFFFFFull);                  // dim0_stride[31:0]
    g1[6] = (int)((st0 >> 32) & 0xFFFFull);              // dim0_stride[47:32]; dim1_stride lo=0
    g1[7] = 0;
    v4i gz = {0, 0, 0, 0};
#if __clang_major__ >= 23
    v8i gz8 = {0, 0, 0, 0, 0, 0, 0, 0};
    __builtin_amdgcn_tensor_load_to_lds(g0, g1, gz, gz, gz8, 0);
#else
    __builtin_amdgcn_tensor_load_to_lds(g0, g1, gz, gz, 0);
#endif
    __builtin_amdgcn_s_wait_tensorcnt(0);
    __syncthreads();
    unsigned int acc = 0;
    for (int i = threadIdx.x; i < 32 * 64; i += 32) acc ^= tileS[i];
    if ((int)__builtin_amdgcn_cluster_id_x() == 0)       // 0 when not in a cluster
        sink[threadIdx.x] = acc;
}
#endif

// ---------------------------------------------------------------------------
// Host side
// ---------------------------------------------------------------------------
extern "C" void kernel_launch(void* const* d_in, const int* in_sizes, int n_in,
                              void* d_out, int out_size, void* d_ws, size_t ws_size,
                              hipStream_t stream) {
    (void)in_sizes; (void)n_in; (void)out_size; (void)ws_size;
    const float* cxyz  = (const float*)d_in[0];
    const float* W_xyz = (const float*)d_in[1];
    const float* b_xyz = (const float*)d_in[2];
    const float* W_lat = (const float*)d_in[3];
    const float* b_lat = (const float*)d_in[4];
    const float* Wq = (const float*)d_in[5];
    const float* Wk = (const float*)d_in[6];
    const float* Wv = (const float*)d_in[7];
    const float* Wo = (const float*)d_in[8];
    const float* bq = (const float*)d_in[9];
    const float* bk = (const float*)d_in[10];
    const float* bv = (const float*)d_in[11];
    const float* bo = (const float*)d_in[12];
    const float* Wf1 = (const float*)d_in[13];
    const float* bf1 = (const float*)d_in[14];
    const float* Wf2 = (const float*)d_in[15];
    const float* bf2 = (const float*)d_in[16];
    const float* g1  = (const float*)d_in[17];
    const float* be1 = (const float*)d_in[18];
    const float* g2  = (const float*)d_in[19];
    const float* be2 = (const float*)d_in[20];
    const float* bWp[16]; const float* bbp[16];
    for (int i = 0; i < 16; ++i) { bWp[i] = (const float*)d_in[21 + i]; bbp[i] = (const float*)d_in[37 + i]; }

    // ---- workspace carve ----
    char* base = (char*)d_ws; size_t off = 0;
    auto alloc = [&](size_t bytes) -> void* {
        void* p = base + off; off = (off + bytes + 255) & ~(size_t)255; return p;
    };
    const size_t BD = (size_t)BATCH * DMODEL;
    unsigned short* latbf   = (unsigned short*)alloc((size_t)BATCH * 1536 * 2);
    unsigned short* xppad   = (unsigned short*)alloc((size_t)BATCH * 32 * 2);
    unsigned short* WxyzB   = (unsigned short*)alloc(32 * 512 * 2);
    unsigned short* WlatB   = (unsigned short*)alloc((size_t)1536 * 512 * 2);
    unsigned short* WqB     = (unsigned short*)alloc(512 * 512 * 2);
    unsigned short* WkB     = (unsigned short*)alloc(512 * 512 * 2);
    unsigned short* WvB     = (unsigned short*)alloc(512 * 512 * 2);
    unsigned short* WoB     = (unsigned short*)alloc(512 * 512 * 2);
    unsigned short* Wf1B    = (unsigned short*)alloc((size_t)512 * 2048 * 2);
    unsigned short* Wf2B    = (unsigned short*)alloc((size_t)2048 * 512 * 2);
    unsigned short* bWB[15];
    for (int i = 0; i < 15; ++i) {
        int kin = (i == 5 || i == 10) ? 1024 : 512;   // first layer of blocks 1,2
        bWB[i] = (unsigned short*)alloc((size_t)kin * 512 * 2);
    }
    float*          xpF     = (float*)alloc(BD * 4);
    unsigned short* xpB     = (unsigned short*)alloc(BD * 2);
    unsigned short* lpB     = (unsigned short*)alloc(BD * 2);
    unsigned short* qB      = (unsigned short*)alloc(BD * 2);
    unsigned short* kB      = (unsigned short*)alloc(BD * 2);
    unsigned short* vB      = (unsigned short*)alloc(BD * 2);
    unsigned short* oB      = (unsigned short*)alloc(BD * 2);
    float*          oprojF  = (float*)alloc(BD * 4);
    float*          x1F     = (float*)alloc(BD * 4);
    unsigned short* x1B     = (unsigned short*)alloc(BD * 2);
    unsigned short* h1B     = (unsigned short*)alloc((size_t)BATCH * 2048 * 2);
    float*          ffF     = (float*)alloc(BD * 4);
    float*          x2F     = (float*)alloc(BD * 4);
    unsigned short* x2B     = (unsigned short*)alloc(BD * 2);
    unsigned short* cat0    = (unsigned short*)alloc((size_t)BATCH * 1024 * 2);
    unsigned short* cat1    = (unsigned short*)alloc((size_t)BATCH * 1024 * 2);
    unsigned short* t0B     = (unsigned short*)alloc(BD * 2);
    unsigned short* t1B     = (unsigned short*)alloc(BD * 2);
    float*          blkF    = (float*)alloc(BD * 4);
    unsigned int*   tdms    = (unsigned int*)alloc(4096);

    auto cvt = [&](const float* s, unsigned short* d, int n) {
        cvt_bf16_kernel<<<(n + 255) / 256, 256, 0, stream>>>(s, d, n);
    };
    auto gemm = [&](const unsigned short* A, int lda, const unsigned short* Bm, int ldb,
                    const float* bias, float* Cf, int ldcf, unsigned short* Cb, int ldcb,
                    int M, int N, int Kd, int act) {
        dim3 grid(N / 64, M / 128);
        gemm_bf16_kernel<<<grid, 256, 0, stream>>>(A, lda, Bm, ldb, bias, Cf, ldcf, Cb, ldcb, Kd, act);
    };

    // ---- weight / input conversions ----
    cvt_lat_kernel<<<(BATCH * 1536) / 256, 256, 0, stream>>>(cxyz, latbf);
    pad_wxyz_kernel<<<(32 * 512) / 256, 256, 0, stream>>>(W_xyz, WxyzB);
    cvt(W_lat, WlatB, 1536 * 512);
    cvt(Wq, WqB, 512 * 512); cvt(Wk, WkB, 512 * 512);
    cvt(Wv, WvB, 512 * 512); cvt(Wo, WoB, 512 * 512);
    cvt(Wf1, Wf1B, 512 * 2048); cvt(Wf2, Wf2B, 2048 * 512);
    for (int i = 0; i < 15; ++i) {
        int kin = (i == 5 || i == 10) ? 1024 : 512;
        cvt(bWp[i], bWB[i], kin * 512);
    }
#ifdef HAVE_TDM
    tdm_stage_kernel<<<1, 32, 0, stream>>>(WlatB, tdms, 512, 1536);
#else
    (void)tdms;
#endif

    // ---- KNN + xp projection ----
    topk_build_kernel<<<BATCH, 32, 0, stream>>>(cxyz, xppad);
    gemm(xppad, 32, WxyzB, 512, b_xyz, xpF, 512, xpB, 512, BATCH, 512, 32, 0);
    // ---- lat projection ----
    gemm(latbf, 1536, WlatB, 512, b_lat, nullptr, 0, lpB, 512, BATCH, 512, 1536, 0);
    // ---- q, k, v ----
    gemm(xpB, 512, WqB, 512, bq, nullptr, 0, qB, 512, BATCH, 512, 512, 0);
    gemm(lpB, 512, WkB, 512, bk, nullptr, 0, kB, 512, BATCH, 512, 512, 0);
    gemm(lpB, 512, WvB, 512, bv, nullptr, 0, vB, 512, BATCH, 512, 512, 0);
    // ---- flash attention (8 waves share K/V tile) ----
    { dim3 grid(BATCH / 128, NHEAD); attn_kernel<<<grid, 256, 0, stream>>>(qB, kB, vB, oB); }
    // ---- output projection + LN1 ----
    gemm(oB, 512, WoB, 512, bo, oprojF, 512, nullptr, 0, BATCH, 512, 512, 0);
    add_ln_kernel<<<BATCH, 32, 0, stream>>>(oprojF, xpF, g1, be1, x1F, x1B);
    // ---- FFN + LN2 ----
    gemm(x1B, 512, Wf1B, 2048, bf1, nullptr, 0, h1B, 2048, BATCH, 2048, 512, 1);
    gemm(h1B, 2048, Wf2B, 512, bf2, ffF, 512, nullptr, 0, BATCH, 512, 2048, 0);
    add_ln_kernel<<<BATCH, 32, 0, stream>>>(ffF, x1F, g2, be2, x2F, x2B);
    // x2 into the concat halves (cols 512..1023)
    copy_off_kernel<<<(BATCH * 512) / 256, 256, 0, stream>>>(x2F, cat0, 1024, 512);
    copy_off_kernel<<<(BATCH * 512) / 256, 256, 0, stream>>>(x2F, cat1, 1024, 512);

    // ---- block 0: 5x relu (512->512), last into cat0[:,0:512] ----
    gemm(x2B, 512, bWB[0], 512, bbp[0], nullptr, 0, t0B, 512, BATCH, 512, 512, 1);
    gemm(t0B, 512, bWB[1], 512, bbp[1], nullptr, 0, t1B, 512, BATCH, 512, 512, 1);
    gemm(t1B, 512, bWB[2], 512, bbp[2], nullptr, 0, t0B, 512, BATCH, 512, 512, 1);
    gemm(t0B, 512, bWB[3], 512, bbp[3], nullptr, 0, t1B, 512, BATCH, 512, 512, 1);
    gemm(t1B, 512, bWB[4], 512, bbp[4], nullptr, 0, cat0, 1024, BATCH, 512, 512, 1);
    // ---- block 1: (1024->512) + 4x(512->512), last into cat1[:,0:512] ----
    gemm(cat0, 1024, bWB[5], 512, bbp[5], nullptr, 0, t0B, 512, BATCH, 512, 1024, 1);
    gemm(t0B, 512, bWB[6], 512, bbp[6], nullptr, 0, t1B, 512, BATCH, 512, 512, 1);
    gemm(t1B, 512, bWB[7], 512, bbp[7], nullptr, 0, t0B, 512, BATCH, 512, 512, 1);
    gemm(t0B, 512, bWB[8], 512, bbp[8], nullptr, 0, t1B, 512, BATCH, 512, 512, 1);
    gemm(t1B, 512, bWB[9], 512, bbp[9], nullptr, 0, cat1, 1024, BATCH, 512, 512, 1);
    // ---- block 2: (1024->512) + 4x(512->512) relu, then 512->1 tanh ----
    gemm(cat1, 1024, bWB[10], 512, bbp[10], nullptr, 0, t0B, 512, BATCH, 512, 1024, 1);
    gemm(t0B, 512, bWB[11], 512, bbp[11], nullptr, 0, t1B, 512, BATCH, 512, 512, 1);
    gemm(t1B, 512, bWB[12], 512, bbp[12], nullptr, 0, t0B, 512, BATCH, 512, 512, 1);
    gemm(t0B, 512, bWB[13], 512, bbp[13], nullptr, 0, t1B, 512, BATCH, 512, 512, 1);
    gemm(t1B, 512, bWB[14], 512, bbp[14], blkF, 512, nullptr, 0, BATCH, 512, 512, 1);
    final_dot_kernel<<<BATCH, 32, 0, stream>>>(blkF, bWp[15], bbp[15], (float*)d_out);
}
